// ExpertPool_8366596292698
// MI455X (gfx1250) — matrix operand order repaired and checked by reading, not compiled, not run on hardware
//
#include <hip/hip_runtime.h>
#include <math.h>

// Problem constants (B=8, S=2048 -> T=16384 tokens)
constexpr int T_TOK = 16384;
constexpr int DIM   = 768;
constexpr int NEXP  = 8;
constexpr int IDIM  = 3072;
constexpr int HDIM  = 1536;
constexpr int SLOTS = 2 * T_TOK;   // top-2 routing -> 32768 token-expert rows

typedef __attribute__((ext_vector_type(16))) __bf16 v16bf;
typedef __attribute__((ext_vector_type(8)))  float  v8f;
typedef __attribute__((ext_vector_type(4)))  int    v4i;

union AB32 { v16bf v; uint4 q[2]; };   // 32 bytes: one WMMA A/B fragment per lane

#if __has_builtin(__builtin_amdgcn_global_load_async_to_lds_b128)
#define USE_ASYNC_LDS 1
#else
#define USE_ASYNC_LDS 0
#endif

typedef __attribute__((address_space(1))) v4i* g4p;   // global int4*
typedef __attribute__((address_space(3))) v4i* l4p;   // LDS int4*

__device__ __forceinline__ void wait_async0() {
#if __has_builtin(__builtin_amdgcn_s_wait_asynccnt)
  __builtin_amdgcn_s_wait_asynccnt(0);
#else
  asm volatile("s_wait_asynccnt 0x0" ::: "memory");
#endif
}

__device__ __forceinline__ unsigned short f2bf(float f) {
  unsigned int u = __float_as_uint(f);
  u += 0x7fffu + ((u >> 16) & 1u);     // round-to-nearest-even
  return (unsigned short)(u >> 16);
}

__device__ __forceinline__ float gelu_erf(float x) {
  return 0.5f * x * (1.0f + erff(x * 0.70710678118654752f));
}

__device__ __forceinline__ int imin(int a, int b) { return a < b ? a : b; }

// ---------------------------------------------------------------- utilities
__global__ void zero_f32_kernel(float* __restrict__ p, int n) {
  for (int i = blockIdx.x * blockDim.x + threadIdx.x; i < n; i += gridDim.x * blockDim.x)
    p[i] = 0.0f;
}

__global__ void zero_i32_kernel(int* __restrict__ p, int n) {
  int i = blockIdx.x * blockDim.x + threadIdx.x;
  if (i < n) p[i] = 0;
}

__global__ void cvt_bf16_kernel(const float* __restrict__ s, unsigned short* __restrict__ d, int n) {
  for (int i = blockIdx.x * blockDim.x + threadIdx.x; i < n; i += gridDim.x * blockDim.x)
    d[i] = f2bf(s[i]);
}

// Transpose-convert: src f32 [E][Kd][Nd] -> dst bf16 [E][Nd][Kd]
// One-time cost; makes GEMM B tiles load exactly like A tiles (no in-loop transpose).
__global__ __launch_bounds__(256)
void cvt_transpose_kernel(const float* __restrict__ src, unsigned short* __restrict__ dst,
                          int Kd, int Nd) {
  __shared__ unsigned short tile[32][33];
  const int e  = blockIdx.z;
  const int k0 = blockIdx.x * 32;
  const int n0 = blockIdx.y * 32;
  const float* s      = src + (size_t)e * Kd * Nd;
  unsigned short* d   = dst + (size_t)e * Nd * Kd;
  const int tx = threadIdx.x & 31;
  const int ty = threadIdx.x >> 5;           // 0..7
#pragma unroll
  for (int i = 0; i < 32; i += 8)            // read coalesced along N
    tile[ty + i][tx] = f2bf(s[(size_t)(k0 + ty + i) * Nd + n0 + tx]);
  __syncthreads();
#pragma unroll
  for (int i = 0; i < 32; i += 8)            // write coalesced along K
    d[(size_t)(n0 + ty + i) * Kd + k0 + tx] = tile[tx][ty + i];
}

// ---------------------------------------------------------------- router
__global__ __launch_bounds__(256)
void router_kernel(const float* __restrict__ x, const float* __restrict__ Wr,
                   const float* __restrict__ br,
                   int* __restrict__ counts, int* __restrict__ sel, float* __restrict__ selw) {
  __shared__ float sW[DIM * NEXP];                 // 24 KB
  for (int i = threadIdx.x; i < DIM * NEXP; i += 256) sW[i] = Wr[i];
  __syncthreads();

  int t = blockIdx.x * 256 + threadIdx.x;
  if (t >= T_TOK) return;

  float logit[NEXP];
#pragma unroll
  for (int e = 0; e < NEXP; ++e) logit[e] = br[e];
  const float* xr = x + (size_t)t * DIM;
  for (int k = 0; k < DIM; ++k) {
    float xv = xr[k];
#pragma unroll
    for (int e = 0; e < NEXP; ++e) logit[e] += xv * sW[k * NEXP + e];
  }
  // top-2 (ties: earliest index first, matching lax.top_k)
  float v0 = -INFINITY, v1 = -INFINITY; int i0 = 0, i1 = 0;
#pragma unroll
  for (int e = 0; e < NEXP; ++e) {
    float v = logit[e];
    if (v > v0)      { v1 = v0; i1 = i0; v0 = v; i0 = e; }
    else if (v > v1) { v1 = v;  i1 = e; }
  }
  float e1 = __expf(v1 - v0);      // v0 >= v1
  float inv = 1.0f / (1.0f + e1);
  atomicAdd(&counts[i0], 1);
  atomicAdd(&counts[i1], 1);
  sel[2 * t]     = i0;  selw[2 * t]     = inv;
  sel[2 * t + 1] = i1;  selw[2 * t + 1] = e1 * inv;
}

// counts at cnt[0..7], bases at cnt[8..15], cursors at cnt[16..23]
__global__ void scan_kernel(int* __restrict__ cnt) {
  if (threadIdx.x == 0) {
    int run = 0;
    for (int e = 0; e < NEXP; ++e) { cnt[8 + e] = run; cnt[16 + e] = run; run += cnt[e]; }
  }
}

__global__ __launch_bounds__(256)
void scatter_kernel(const int* __restrict__ sel, const float* __restrict__ selw,
                    int* __restrict__ cnt, int* __restrict__ rowtok, float* __restrict__ roww) {
  int i = blockIdx.x * 256 + threadIdx.x;
  if (i >= SLOTS) return;
  int e   = sel[i];
  int pos = atomicAdd(&cnt[16 + e], 1);   // cursor starts at bases[e]
  rowtok[pos] = i >> 1;
  roww[pos]   = selw[i];
}

// ---------------------------------------------------------------- grouped GEMM
// One expert per blockIdx.z; 128x128 block tile, BK=64, double-buffered LDS.
// A: bf16 [rows][K] (optionally token-indirected). B: bf16 [E][N][K] (pre-transposed)
// so both tiles are straight memory->LDS copies -> GLOBAL_LOAD_ASYNC_TO_LDS_B128
// (zero VGPR staging, ASYNCcnt-tracked, overlapped with WMMA).
// 8 waves; each wave computes 32x64 via 2x4 v_wmma_f32_16x16x32_bf16 (16 WMMA
// per k-tile, one barrier per k-tile). MODE 0: C = bf16(gelu(acc+bias)).
// MODE 1: out[token,*] += combine_w * (acc+bias)  (atomic scatter).
template <int N, int K, int MODE, bool INDIR>
__global__ __launch_bounds__(256, 1)
void moe_gemm(const unsigned short* __restrict__ Abase,   // bf16; [rows,K] or x[T,K] if INDIR
              const unsigned short* __restrict__ Btbase,  // bf16; [E,N,K] pre-transposed
              const float* __restrict__ bias,             // [E,N]
              unsigned short* __restrict__ Cbf,           // MODE 0 output (bf16)
              float* __restrict__ Out,                    // MODE 1 output (f32, [T,DIM])
              const int* __restrict__ counts, const int* __restrict__ bases,
              const int* __restrict__ rowtok, const float* __restrict__ roww) {
  constexpr int BM = 128, BN = 128, BK = 64, LDT = BK + 8;  // +8 ushorts padding
  constexpr int TILE = BM * LDT;                            // ushorts per buffer
  __shared__ unsigned short sA[2 * TILE];   // [m][k], double-buffered
  __shared__ unsigned short sB[2 * TILE];   // [n][k], double-buffered

  const int e   = blockIdx.z;
  const int cnt = counts[e];
  const int m0  = blockIdx.x * BM;
  if (m0 >= cnt) return;                    // uniform early-exit (EXEC stays all-1s)
  const int mrows = imin(BM, cnt - m0);
  const int base  = bases[e];
  const int n0    = blockIdx.y * BN;

  const unsigned short* Bexp = Btbase + (size_t)e * N * K;  // uniform

  const int tid  = threadIdx.x;
  const int lane = tid & 31;
  const int wid  = tid >> 5;
  const int wm   = wid & 3;                 // 0..3 -> 32-row span
  const int wn   = wid >> 2;                // 0..1 -> 64-col span

  // ---- per-thread load geometry (32-bit offsets; token indirection hoisted)
  int aoff[4], boff[4], soff[4];
#pragma unroll
  for (int it = 0; it < 4; ++it) {
    int c  = tid + it * 256;                // 0..1023
    int r  = c >> 3;                        // tile row 0..127 (8 x 16B chunks per row)
    int kc = (c & 7) * 8;                   // col chunk (8 ushorts = 16B)
    int ml = imin(r, mrows - 1);
    if (INDIR) {
      int tok  = rowtok[base + m0 + ml];
      aoff[it] = tok * K + kc;
    } else {
      aoff[it] = (base + m0 + ml) * K + kc;
    }
    boff[it] = (n0 + r) * K + kc;           // B rows are N (pre-transposed layout)
    soff[it] = r * LDT + kc;                // same LDS placement for A and B
  }

#if USE_ASYNC_LDS
  // Issue 8 async 16B copies (4 A + 4 B) into LDS buffer `buf` for k-tile k0.
  auto issue_tiles = [&](int buf, int k0) {
    unsigned short* dA = sA + buf * TILE;
    unsigned short* dB = sB + buf * TILE;
#pragma unroll
    for (int it = 0; it < 4; ++it) {
      __builtin_amdgcn_global_load_async_to_lds_b128(
          (g4p)(Abase + aoff[it] + k0), (l4p)(dA + soff[it]), 0, 0);
      __builtin_amdgcn_global_load_async_to_lds_b128(
          (g4p)(Bexp + boff[it] + k0), (l4p)(dB + soff[it]), 0, 0);
    }
  };
#else
  uint4 ra[4], rb[4];                       // register staging fallback
  auto load_regs = [&](int k0) {
#pragma unroll
    for (int it = 0; it < 4; ++it) {
      ra[it] = *(const uint4*)(Abase + aoff[it] + k0);
      rb[it] = *(const uint4*)(Bexp + boff[it] + k0);
    }
  };
  auto store_tiles = [&](int buf) {
    unsigned short* dA = sA + buf * TILE;
    unsigned short* dB = sB + buf * TILE;
#pragma unroll
    for (int it = 0; it < 4; ++it) {
      *(uint4*)(dA + soff[it]) = ra[it];
      *(uint4*)(dB + soff[it]) = rb[it];
    }
  };
#endif

  const v8f vzero = {0.f, 0.f, 0.f, 0.f, 0.f, 0.f, 0.f, 0.f};
  v8f acc[2][4];
#pragma unroll
  for (int im = 0; im < 2; ++im)
#pragma unroll
    for (int in = 0; in < 4; ++in) acc[im][in] = vzero;

  // B fragments streamed one at a time (each feeds 2 WMMAs) to cap register
  // pressure: peak frag liveness 2xA + 1xB = 24 VGPRs instead of 48.
  auto compute = [&](int buf) {
    const unsigned short* cA = sA + buf * TILE;
    const unsigned short* cB = sB + buf * TILE;
#pragma unroll
    for (int ks = 0; ks < BK; ks += 32) {
      AB32 af[2];
      int mrow = wm * 32 + (lane & 15);
      int kh   = ks + ((lane >> 4) * 8);    // lanes<16: K 0-7 & 16-23; else 8-15 & 24-31
#pragma unroll
      for (int im = 0; im < 2; ++im) {
        const unsigned short* p = cA + (mrow + im * 16) * LDT + kh;
        af[im].q[0] = *(const uint4*)p;
        af[im].q[1] = *(const uint4*)(p + 16);
      }
      int nrow = wn * 64 + (lane & 15);
      int koff = ks + ((lane >> 4) * 8);    // lanes<16: K 0-7 & 16-23; else 8-15 & 24-31
#pragma unroll
      for (int in = 0; in < 4; ++in) {
        AB32 bf;
        const unsigned short* p = cB + (nrow + in * 16) * LDT + koff;
        bf.q[0] = *(const uint4*)p;
        bf.q[1] = *(const uint4*)(p + 16);
        acc[0][in] = __builtin_amdgcn_wmma_f32_16x16x32_bf16(
            false, af[0].v, false, bf.v, (short)0, acc[0][in], false, false);
        acc[1][in] = __builtin_amdgcn_wmma_f32_16x16x32_bf16(
            false, af[1].v, false, bf.v, (short)0, acc[1][in], false, false);
      }
    }
  };

  // ---- double-buffered mainloop: one barrier per 64-deep k-tile (16 WMMAs)
  constexpr int NT = K / BK;
#if USE_ASYNC_LDS
  issue_tiles(0, 0);
  wait_async0();
  __syncthreads();
  int cur = 0;
  for (int kt = 0; kt < NT; ++kt) {
    const bool has_next = (kt + 1 < NT);    // uniform
    if (has_next) issue_tiles(cur ^ 1, (kt + 1) * BK);  // DMA overlaps WMMA
    compute(cur);
    if (has_next) wait_async0();
    __syncthreads();
    cur ^= 1;
  }
#else
  load_regs(0);
  store_tiles(0);
  __syncthreads();
  int cur = 0;
  for (int kt = 0; kt < NT; ++kt) {
    const bool has_next = (kt + 1 < NT);    // uniform
    if (has_next) load_regs((kt + 1) * BK);
    compute(cur);
    if (has_next) store_tiles(cur ^ 1);
    __syncthreads();
    cur ^= 1;
  }
#endif

  // ---- epilogue
  const float* be = bias + (size_t)e * N;
#pragma unroll
  for (int im = 0; im < 2; ++im) {
#pragma unroll
    for (int in = 0; in < 4; ++in) {
      int mr0 = wm * 32 + im * 16 + ((lane >> 4) * 8);
      int col = n0 + wn * 64 + in * 16 + (lane & 15);
      float bv = be[col];
#pragma unroll
      for (int r = 0; r < 8; ++r) {
        int mr = mr0 + r;
        if (mr < mrows) {
          float v = acc[im][in][r] + bv;
          if (MODE == 0) {
            Cbf[(size_t)(base + m0 + mr) * N + col] = f2bf(gelu_erf(v));
          } else {
            int slot = base + m0 + mr;
            int tok  = rowtok[slot];
            float w  = roww[slot];
            atomicAdd(&Out[(size_t)tok * DIM + col], w * v);
          }
        }
      }
    }
  }
}

// ---------------------------------------------------------------- launcher
extern "C" void kernel_launch(void* const* d_in, const int* in_sizes, int n_in,
                              void* d_out, int out_size, void* d_ws, size_t ws_size,
                              hipStream_t stream) {
  (void)in_sizes; (void)n_in; (void)out_size; (void)ws_size;
  const float* x  = (const float*)d_in[0];
  const float* Wr = (const float*)d_in[1];
  const float* br = (const float*)d_in[2];
  const float* W1 = (const float*)d_in[3];
  const float* b1 = (const float*)d_in[4];
  const float* W2 = (const float*)d_in[5];
  const float* b2 = (const float*)d_in[6];
  const float* W3 = (const float*)d_in[7];
  const float* b3 = (const float*)d_in[8];
  // d_in[9] = top_k, fixed at 2 here.
  float* out = (float*)d_out;

  // Workspace carve-up (~460 MB total), 256B aligned slices.
  char* ws = (char*)d_ws;
  size_t off = 0;
  auto carve = [&](size_t bytes) -> void* {
    void* p = ws + off;
    off = (off + bytes + 255) & ~(size_t)255;
    return p;
  };
  unsigned short* xb  = (unsigned short*)carve((size_t)T_TOK * DIM * 2);
  unsigned short* w1t = (unsigned short*)carve((size_t)NEXP * DIM * IDIM * 2);  // [E][I][D]
  unsigned short* w2t = (unsigned short*)carve((size_t)NEXP * IDIM * HDIM * 2); // [E][H][I]
  unsigned short* w3t = (unsigned short*)carve((size_t)NEXP * HDIM * DIM * 2);  // [E][D][H]
  int*            cnt = (int*)carve(24 * sizeof(int));      // counts|bases|cursor
  int*            sel = (int*)carve((size_t)SLOTS * 4);
  float*         selw = (float*)carve((size_t)SLOTS * 4);
  int*         rowtok = (int*)carve((size_t)SLOTS * 4);
  float*         roww = (float*)carve((size_t)SLOTS * 4);
  unsigned short*  h1 = (unsigned short*)carve((size_t)SLOTS * IDIM * 2);
  unsigned short*  h2 = (unsigned short*)carve((size_t)SLOTS * HDIM * 2);

  // 1) init
  zero_f32_kernel<<<2048, 256, 0, stream>>>(out, T_TOK * DIM);
  zero_i32_kernel<<<1, 32, 0, stream>>>(cnt, 24);

  // 2) f32 -> bf16; weights also transposed to [E][N][K] (one-time, off critical path)
  cvt_bf16_kernel<<<4096, 256, 0, stream>>>(x, xb, T_TOK * DIM);
  cvt_transpose_kernel<<<dim3(DIM / 32, IDIM / 32, NEXP), 256, 0, stream>>>(W1, w1t, DIM, IDIM);
  cvt_transpose_kernel<<<dim3(IDIM / 32, HDIM / 32, NEXP), 256, 0, stream>>>(W2, w2t, IDIM, HDIM);
  cvt_transpose_kernel<<<dim3(HDIM / 32, DIM / 32, NEXP), 256, 0, stream>>>(W3, w3t, HDIM, DIM);

  // 3) routing
  router_kernel<<<T_TOK / 256, 256, 0, stream>>>(x, Wr, br, cnt, sel, selw);
  scan_kernel<<<1, 32, 0, stream>>>(cnt);
  scatter_kernel<<<SLOTS / 256, 256, 0, stream>>>(sel, selw, cnt, rowtok, roww);

  // 4) grouped expert MLP (worst-case grids; blocks exit on per-expert counts)
  moe_gemm<IDIM, DIM, 0, true><<<dim3(T_TOK / 128, IDIM / 128, NEXP), 256, 0, stream>>>(
      xb, w1t, b1, h1, nullptr, cnt, cnt + 8, rowtok, roww);
  moe_gemm<HDIM, IDIM, 0, false><<<dim3(T_TOK / 128, HDIM / 128, NEXP), 256, 0, stream>>>(
      h1, w2t, b2, h2, nullptr, cnt, cnt + 8, rowtok, roww);
  moe_gemm<DIM, HDIM, 1, false><<<dim3(T_TOK / 128, DIM / 128, NEXP), 256, 0, stream>>>(
      h2, w3t, b3, nullptr, out, cnt, cnt + 8, rowtok, roww);
}
